// IterSpatialCorrelationSampler_7885559955777
// MI455X (gfx1250) — matrix operand order repaired and checked by reading, not compiled, not run on hardware
//
#include <hip/hip_runtime.h>

typedef __attribute__((ext_vector_type(16))) __bf16 v16bf;
typedef __attribute__((ext_vector_type(8)))  __bf16 v8bf;
typedef __attribute__((ext_vector_type(8)))  float  v8f;

#define BB   4
#define CC   256
#define HH   96
#define WW   128
#define PP   9
#define OFF  4            // (PP-1)/2
#define KCH  32           // K per WMMA (bf16)
#define NKC  (CC / KCH)   // 8 K-chunks
#define WPB  4            // waves per block (main kernel)

#define HP   (HH + 2 * OFF)          // 104 padded rows
#define WP   (WW + 2 * OFF + 8)      // 144 padded cols (tile1 reach: w0+27 -> wp 143)
#define A_ELEMS ((long)BB * HH * WW * CC)
#define B_ELEMS ((long)BB * HP * WP * CC)

// ---------------------------------------------------------------------------
// Pack in1 -> bf16 [b][h][w][c] (channel-contiguous). One wave per pixel,
// lane L handles channels 8L..8L+7; stores one contiguous b128 per lane.
// ---------------------------------------------------------------------------
__global__ __launch_bounds__(256)
void pack_in1(const float* __restrict__ in1, __bf16* __restrict__ pA)
{
    const int  p    = __builtin_amdgcn_readfirstlane(blockIdx.x * 8 + (threadIdx.x >> 5));
    const int  lane = threadIdx.x & 31;
    const long hw   = (long)HH * WW;

    const int b  = p / (HH * WW);
    const int ph = p % (HH * WW);
    const float* src = in1 + (long)b * CC * hw + ph;   // + c*hw

    v8bf v;
#pragma unroll
    for (int i = 0; i < 8; ++i)
        v[i] = (__bf16)src[(long)(lane * 8 + i) * hw];

    *(v8bf*)(pA + (long)p * CC + lane * 8) = v;
}

// ---------------------------------------------------------------------------
// Pack in2 -> zero-padded bf16 [b][hp=HP][wp=WP][c]. Halo rows/cols zeroed.
// ---------------------------------------------------------------------------
__global__ __launch_bounds__(256)
void pack_in2(const float* __restrict__ in2, __bf16* __restrict__ pB)
{
    const int  p    = __builtin_amdgcn_readfirstlane(blockIdx.x * 8 + (threadIdx.x >> 5));
    const int  lane = threadIdx.x & 31;
    const long hw   = (long)HH * WW;

    const int b   = p / (HP * WP);
    const int hp  = (p / WP) % HP;
    const int wpx = p % WP;
    const int row = hp  - OFF;
    const int col = wpx - OFF;

    v8bf v;
    if (row >= 0 && row < HH && col >= 0 && col < WW) {   // wave-uniform
        const float* src = in2 + (long)b * CC * hw + (long)row * WW + col;
#pragma unroll
        for (int i = 0; i < 8; ++i)
            v[i] = (__bf16)src[(long)(lane * 8 + i) * hw];
    } else {
#pragma unroll
        for (int i = 0; i < 8; ++i) v[i] = (__bf16)0.0f;
    }

    *(v8bf*)(pB + (long)p * CC + lane * 8) = v;
}

// ---------------------------------------------------------------------------
// Main WMMA kernel on packed bf16: one wave per 16-pixel strip, 18 bf16 WMMA
// chains (9 dy x 2 horizontal B-tiles), K = C = 256. No masks, no converts:
// each fragment is two contiguous b128 loads.
// ---------------------------------------------------------------------------
__global__ __launch_bounds__(WPB * 32)
void corr_wmma_packed(const __bf16* __restrict__ pA,
                      const __bf16* __restrict__ pB,
                      float* __restrict__ out)
{
    const int s  = __builtin_amdgcn_readfirstlane(blockIdx.x * WPB + (threadIdx.x >> 5));
    const int wt = s & 7;                 // WW/16 == 8 strips per row
    const int h  = (s >> 3) % HH;
    const int b  = s / (8 * HH);
    const int w0 = wt * 16;

    const int lane = threadIdx.x & 31;
    const int half = lane >> 4;
    const int n    = lane & 15;           // N column (B/C/D); also A row index

    v8f acc[2 * PP];
#pragma unroll
    for (int i = 0; i < 2 * PP; ++i)
        acc[i] = v8f{0.f,0.f,0.f,0.f,0.f,0.f,0.f,0.f};

    // Scalar bases (SGPR) + per-lane dword offsets (VGPR); loop/dy offsets
    // become immediates.
    const __bf16* baseA = pA + (long)((b * HH + h) * WW) * CC;
    const __bf16* baseB = pB + (long)((b * HP + h) * WP) * CC;   // hp = h + dy
    const int voffA = (w0 + n) * CC + 16 * half;
    int voffB[2];
#pragma unroll
    for (int t = 0; t < 2; ++t)
        voffB[t] = (w0 + t * 16 + n) * CC + 16 * half;

#pragma unroll 1
    for (int kc = 0; kc < NKC; ++kc) {
        const int kb = kc * KCH;          // channel offset of this K-chunk

        // A 16x32 bf16: element j -> K = kb + j + 16*half (16 contiguous).
        v8bf a0 = *(const v8bf*)(baseA + voffA + kb);
        v8bf a1 = *(const v8bf*)(baseA + voffA + kb + 8);
        v16bf a = __builtin_shufflevector(a0, a1, 0,1,2,3,4,5,6,7,8,9,10,11,12,13,14,15);

#pragma unroll
        for (int dy = 0; dy < PP; ++dy) {
#pragma unroll
            for (int t = 0; t < 2; ++t) {
                // B 32x16 bf16: element j -> K = kb + j + 16*half (16 contiguous).
                const __bf16* q = baseB + (long)dy * WP * CC + voffB[t] + kb;
                v8bf b0 = *(const v8bf*)(q);
                v8bf b1 = *(const v8bf*)(q + 8);
                v16bf bv = __builtin_shufflevector(b0, b1, 0,1,2,3,4,5,6,7,8,9,10,11,12,13,14,15);

                acc[dy * 2 + t] = __builtin_amdgcn_wmma_f32_16x16x32_bf16(
                    false, a, false, bv, (short)0, acc[dy * 2 + t], false, false);
            }
        }
    }

    // Scatter D to out[b, dy, dx+4, h, w0+m]; VGPR r holds (M=r+8*half, N=n).
    // tile0: dx = n-m-4 ; tile1: dx = n-m+12 ; disjoint, cover [-4,4] once.
    const long hw = (long)HH * WW;
    float* po = out + ((long)b * (PP * PP) * HH + h) * WW;
#pragma unroll
    for (int dy = 0; dy < PP; ++dy) {
#pragma unroll
        for (int t = 0; t < 2; ++t) {
#pragma unroll
            for (int r = 0; r < 8; ++r) {
                int m  = r + 8 * half;
                int dx = (t == 0) ? (n - m - OFF) : (n - m + 12);
                if (dx >= -OFF && dx <= OFF) {
                    int q = dy * PP + (dx + OFF);
                    po[(long)q * hw + (w0 + m)] = acc[dy * 2 + t][r];
                }
            }
        }
    }
}

// ---------------------------------------------------------------------------
// Fallback (no workspace): direct-from-global version (round-2 kernel).
// ---------------------------------------------------------------------------
__global__ __launch_bounds__(WPB * 32)
void corr_wmma_direct(const float* __restrict__ in1,
                      const float* __restrict__ in2,
                      float* __restrict__ out)
{
    const int s  = __builtin_amdgcn_readfirstlane(blockIdx.x * WPB + (threadIdx.x >> 5));
    const int wt = s & 7;
    const int h  = (s >> 3) % HH;
    const int b  = s / (8 * HH);
    const int w0 = wt * 16;

    const int lane = threadIdx.x & 31;
    const int half = lane >> 4;
    const int n    = lane & 15;

    v8f acc[2 * PP];
#pragma unroll
    for (int i = 0; i < 2 * PP; ++i)
        acc[i] = v8f{0.f,0.f,0.f,0.f,0.f,0.f,0.f,0.f};

    const long hw = (long)HH * WW;
    const float* p1 = in1 + ((long)b * CC * HH + h) * WW + (w0 + n);
    const float* p2 = in2 + (long)b * CC * HH * WW;

    int  colc[2];
    bool colok[2];
#pragma unroll
    for (int t = 0; t < 2; ++t) {
        int col  = w0 - OFF + t * 16 + n;
        colok[t] = (col >= 0) && (col < WW);
        colc[t]  = col < 0 ? 0 : (col >= WW ? WW - 1 : col);
    }

#pragma unroll 1
    for (int kc = 0; kc < NKC; ++kc) {
        const int kbase = kc * KCH;
        const float* qa = p1 + (long)(kbase + 16 * half) * hw;
        v16bf a;
#pragma unroll
        for (int j = 0; j < 16; ++j) {
            int ko = j;
            a[j] = (__bf16)qa[(long)ko * hw];
        }

#pragma unroll
        for (int dy = 0; dy < PP; ++dy) {
            const int row = h + dy - OFF;
            if (row >= 0 && row < HH) {
#pragma unroll
                for (int t = 0; t < 2; ++t) {
                    const float* qb = p2 + (long)(kbase + 16 * half) * hw
                                         + (long)row * WW + colc[t];
                    v16bf bv;
#pragma unroll
                    for (int j = 0; j < 16; ++j) {
                        float f = qb[(long)j * hw];
                        f = colok[t] ? f : 0.0f;
                        bv[j] = (__bf16)f;
                    }
                    acc[dy * 2 + t] = __builtin_amdgcn_wmma_f32_16x16x32_bf16(
                        false, a, false, bv, (short)0, acc[dy * 2 + t], false, false);
                }
            }
        }
    }

    float* po = out + ((long)b * (PP * PP) * HH + h) * WW;
#pragma unroll
    for (int dy = 0; dy < PP; ++dy) {
#pragma unroll
        for (int t = 0; t < 2; ++t) {
#pragma unroll
            for (int r = 0; r < 8; ++r) {
                int m  = r + 8 * half;
                int dx = (t == 0) ? (n - m - OFF) : (n - m + 12);
                if (dx >= -OFF && dx <= OFF) {
                    int q = dy * PP + (dx + OFF);
                    po[(long)q * hw + (w0 + m)] = acc[dy * 2 + t][r];
                }
            }
        }
    }
}

extern "C" void kernel_launch(void* const* d_in, const int* in_sizes, int n_in,
                              void* d_out, int out_size, void* d_ws, size_t ws_size,
                              hipStream_t stream)
{
    (void)in_sizes; (void)n_in; (void)out_size;
    const float* in1 = (const float*)d_in[0];
    const float* in2 = (const float*)d_in[1];
    float* out = (float*)d_out;

    const int strips = BB * HH * (WW / 16);          // 3072 waves
    const size_t need = (size_t)(A_ELEMS + B_ELEMS) * sizeof(__bf16);

    if (ws_size >= need && d_ws != nullptr) {
        __bf16* pA = (__bf16*)d_ws;
        __bf16* pB = pA + A_ELEMS;
        pack_in1<<<(BB * HH * WW) / 8, 256, 0, stream>>>(in1, pA);
        pack_in2<<<(BB * HP * WP) / 8, 256, 0, stream>>>(in2, pB);
        corr_wmma_packed<<<strips / WPB, WPB * 32, 0, stream>>>(pA, pB, out);
    } else {
        corr_wmma_direct<<<strips / WPB, WPB * 32, 0, stream>>>(in1, in2, out);
    }
}
